// MatrixNMS_1468878815721
// MI455X (gfx1250) — 8-bit matrix operand order repaired and checked by reading, compiled, not run on hardware
//
#include <hip/hip_runtime.h>
#include <hip/hip_bf16.h>
#include <stdint.h>


typedef __attribute__((ext_vector_type(16))) int          v16i;
typedef __attribute__((ext_vector_type(8)))  float        v8f;
typedef __attribute__((ext_vector_type(4)))  unsigned int v4u;
typedef __attribute__((ext_vector_type(8)))  int          v8i;
typedef __attribute__((ext_vector_type(4)))  int          v4i;

#define N_CAND 1024
#define HW     40960
#define SIGMA  2.0f
#define FP8_ONE 0x38u
#define LDS_PITCH 144

#if defined(__has_builtin)
#  if __has_builtin(__builtin_amdgcn_tensor_load_to_lds)
#    define HAVE_TDM 1
#  endif
#endif
#ifndef HAVE_TDM
#  define HAVE_TDM 0
#endif

#if __has_include(<hip/amd_detail/amd_gfx1250_TDM.h>)
#  define TDM_6ARG 1
#else
#  define TDM_6ARG 0
#endif

__global__ __launch_bounds__(256) void k_convert(const float* __restrict__ in,
                                                 uint32_t* __restrict__ out8,
                                                 int total4) {
  int idx = blockIdx.x * 256 + threadIdx.x;
  if (idx >= total4) return;
  const float4 f = reinterpret_cast<const float4*>(in)[idx];
  uint32_t b = (f.x != 0.f ? FP8_ONE : 0u)
             | ((f.y != 0.f ? FP8_ONE : 0u) << 8)
             | ((f.z != 0.f ? FP8_ONE : 0u) << 16)
             | ((f.w != 0.f ? FP8_ONE : 0u) << 24);
  out8[idx] = b;
}

typedef uint8_t __attribute__((address_space(3))) lds_u8_t;

__device__ __forceinline__ uint32_t lds_addr_of(const void* p) {
  return (uint32_t)(uintptr_t)(const lds_u8_t*)p;
}

#if HAVE_TDM
__device__ __forceinline__ void tdm_load_2d(uint32_t lds, const uint8_t* gptr,
                                            uint32_t rows) {
  const uint64_t ga = (uint64_t)(uintptr_t)gptr;
  v4u g0;
  g0.x = 1u;
  g0.y = lds;
  g0.z = (uint32_t)ga;
  g0.w = (uint32_t)((ga >> 32) & 0x1FFFFFFu)
       | (2u << 30);
  v8i g1;
  g1[0] = (int)((1u << 20)
              | (4u << 22)
              | (3u << 25));
  g1[1] = (int)((HW & 0xFFFFu) << 16);
  g1[2] = (int)((HW >> 16) | ((N_CAND & 0xFFFFu) << 16));
  g1[3] = (int)(128u << 16);
  g1[4] = (int)rows;
  g1[5] = (int)HW;
  g1[6] = 0;
  g1[7] = 0;
  const v4i z4 = {0, 0, 0, 0};
#if TDM_6ARG
  const v8i z8 = {0, 0, 0, 0, 0, 0, 0, 0};
  __builtin_amdgcn_tensor_load_to_lds(g0, g1, z4, z4, z8, 0);
#else
  __builtin_amdgcn_tensor_load_to_lds(g0, g1, z4, z4, 0);
#endif
}
#else
__device__ __forceinline__ void stage_tile(uint8_t* dst, const uint8_t* src,
                                           int rows, int tid) {
  const int nvec = rows * 8;
  for (int i = tid; i < nvec; i += 256) {
    const int r = i >> 3, c = (i & 7) * 16;
    *reinterpret_cast<uint4*>(dst + r * LDS_PITCH + c) =
        *reinterpret_cast<const uint4*>(src + (size_t)r * HW + c);
  }
}
#endif

union AFrag { v16i v; uint2 q[8]; };
union BFrag { v16i v; uint4 x[4]; };

__global__ __launch_bounds__(256) void k_gemm(const uint8_t* __restrict__ A8,
                                              float* __restrict__ C) {
  __shared__ __align__(16) uint8_t sA[3][64 * LDS_PITCH];
  __shared__ __align__(16) uint8_t sB[3][128 * LDS_PITCH];

  const int lane = threadIdx.x & 31;
  const int wave = threadIdx.x >> 5;
  const int l16  = lane & 15;
  const int half = lane >> 4;
  const int wy   = wave >> 2;
  const int wx   = wave & 3;
  const int rmBlk = blockIdx.x * 64;
  const int cnBlk = blockIdx.y * 128;

  const int ar0 = (wy * 32 + l16)      * LDS_PITCH;
  const int ar1 = (wy * 32 + 16 + l16) * LDS_PITCH;
  const int br0 = (wx * 32 + l16)      * LDS_PITCH;
  const int br1 = (wx * 32 + 16 + l16) * LDS_PITCH;

  v8f acc00 = {}, acc01 = {}, acc10 = {}, acc11 = {};
  const int NK = HW / 128;

#if HAVE_TDM
  if (wave == 0) {
    tdm_load_2d(lds_addr_of(&sA[0][0]), A8 + (size_t)rmBlk * HW, 64);
    tdm_load_2d(lds_addr_of(&sB[0][0]), A8 + (size_t)cnBlk * HW, 128);
    tdm_load_2d(lds_addr_of(&sA[1][0]), A8 + (size_t)rmBlk * HW + 128, 64);
    tdm_load_2d(lds_addr_of(&sB[1][0]), A8 + (size_t)cnBlk * HW + 128, 128);
  }
  int cur = 0;
#endif

#pragma unroll 2
  for (int k = 0; k < NK; ++k) {
#if HAVE_TDM
    if (wave == 0) {
      if (k + 1 < NK) __builtin_amdgcn_s_wait_tensorcnt(2);
      else            __builtin_amdgcn_s_wait_tensorcnt(0);
    }
    __syncthreads();
    if (wave == 0 && k + 2 < NK) {
      const int pf = (cur >= 1) ? cur - 1 : cur + 2;
      tdm_load_2d(lds_addr_of(&sA[pf][0]),
                  A8 + (size_t)rmBlk * HW + (size_t)(k + 2) * 128, 64);
      tdm_load_2d(lds_addr_of(&sB[pf][0]),
                  A8 + (size_t)cnBlk * HW + (size_t)(k + 2) * 128, 128);
    }
    const uint8_t* pa = &sA[cur][0];
    const uint8_t* pb = &sB[cur][0];
    cur = (cur == 2) ? 0 : cur + 1;
#else
    __syncthreads();
    stage_tile(&sA[0][0], A8 + (size_t)rmBlk * HW + (size_t)k * 128, 64,
               threadIdx.x);
    stage_tile(&sB[0][0], A8 + (size_t)cnBlk * HW + (size_t)k * 128, 128,
               threadIdx.x);
    __syncthreads();
    const uint8_t* pa = &sA[0][0];
    const uint8_t* pb = &sB[0][0];
#endif

    AFrag a0, a1; BFrag b0, b1;
#pragma unroll
    for (int qq = 0; qq < 8; ++qq) {
      const int off = 32 * (qq >> 1) + half * 16 + (qq & 1) * 8;
      a0.q[qq] = *reinterpret_cast<const uint2*>(pa + ar0 + off);
      a1.q[qq] = *reinterpret_cast<const uint2*>(pa + ar1 + off);
    }
#pragma unroll
    for (int cc = 0; cc < 4; ++cc) {
      const int off = 32 * cc + half * 16;
      b0.x[cc] = *reinterpret_cast<const uint4*>(pb + br0 + off);
      b1.x[cc] = *reinterpret_cast<const uint4*>(pb + br1 + off);
    }
    acc00 = __builtin_amdgcn_wmma_f32_16x16x128_fp8_fp8(a0.v, b0.v, (short)0, acc00, false, false);
    acc01 = __builtin_amdgcn_wmma_f32_16x16x128_fp8_fp8(a0.v, b1.v, (short)0, acc01, false, false);
    acc10 = __builtin_amdgcn_wmma_f32_16x16x128_fp8_fp8(a1.v, b0.v, (short)0, acc10, false, false);
    acc11 = __builtin_amdgcn_wmma_f32_16x16x128_fp8_fp8(a1.v, b1.v, (short)0, acc11, false, false);
  }

  const int rm = rmBlk + wy * 32;
  const int cn = cnBlk + wx * 32;
  const int mbase = rm + half * 8;
  const int nb0 = cn + l16, nb1 = cn + 16 + l16;
#pragma unroll
  for (int r = 0; r < 8; ++r) {
    C[(size_t)(mbase + r)      * N_CAND + nb0] = acc00[r];
    C[(size_t)(mbase + r)      * N_CAND + nb1] = acc01[r];
    C[(size_t)(mbase + 16 + r) * N_CAND + nb0] = acc10[r];
    C[(size_t)(mbase + 16 + r) * N_CAND + nb1] = acc11[r];
  }
}

__global__ __launch_bounds__(256) void k_comp(const float* __restrict__ C,
                                              const int* __restrict__ labels,
                                              float* __restrict__ comp) {
  int j = blockIdx.x * 256 + threadIdx.x;
  if (j >= N_CAND) return;
  const float sum_j = C[(size_t)j * N_CAND + j];
  const int lab_j = labels[j];
  float m = 0.f;
  for (int i = 0; i < j; ++i) {
    if (labels[i] == lab_j) {
      const float inter = C[(size_t)i * N_CAND + j];
      const float sum_i = C[(size_t)i * N_CAND + i];
      m = fmaxf(m, inter / (sum_i + sum_j - inter));
    }
  }
  comp[j] = m;
}

__global__ __launch_bounds__(256) void k_decay(const float* __restrict__ C,
                                               const int* __restrict__ labels,
                                               const float* __restrict__ comp,
                                               const float* __restrict__ scores,
                                               float* __restrict__ out) {
  int j = blockIdx.x * 256 + threadIdx.x;
  if (j >= N_CAND) return;
  const float sum_j = C[(size_t)j * N_CAND + j];
  const int lab_j = labels[j];
  float mn = 3.402823466e38f;
  for (int i = 0; i < N_CAND; ++i) {
    const float ci = comp[i];
    float d = 0.f;
    if (i < j && labels[i] == lab_j) {
      const float inter = C[(size_t)i * N_CAND + j];
      const float sum_i = C[(size_t)i * N_CAND + i];
      d = inter / (sum_i + sum_j - inter);
    }
    const float term = __expf(SIGMA * (ci * ci - d * d));
    mn = fminf(mn, term);
  }
  out[j] = scores[j] * mn;
}

extern "C" void kernel_launch(void* const* d_in, const int* in_sizes, int n_in,
                              void* d_out, int out_size, void* d_ws, size_t ws_size,
                              hipStream_t stream) {
  const float* seg    = (const float*)d_in[0];
  const int*   labels = (const int*)d_in[1];
  const float* scores = (const float*)d_in[2];
  float* out = (float*)d_out;

  uint8_t* ws   = (uint8_t*)d_ws;
  uint8_t* A8   = ws;
  float*   C    = (float*)(ws + (size_t)N_CAND * HW);
  float*   comp = (float*)(ws + (size_t)N_CAND * HW
                              + (size_t)N_CAND * N_CAND * 4);

  const int total4 = N_CAND * HW / 4;
  k_convert<<<(total4 + 255) / 256, 256, 0, stream>>>(seg, (uint32_t*)A8, total4);
  k_gemm<<<dim3(N_CAND / 64, N_CAND / 128), 256, 0, stream>>>(A8, C);
  k_comp<<<N_CAND / 256, 256, 0, stream>>>(C, labels, comp);
  k_decay<<<N_CAND / 256, 256, 0, stream>>>(C, labels, comp, scores, out);
}
